// SpGraphAttentionLayer_25520695673169
// MI455X (gfx1250) — matrix operand order repaired and checked by reading, not compiled, not run on hardware
//
#include <hip/hip_runtime.h>
#include <hip/hip_bf16.h>

typedef __attribute__((ext_vector_type(16))) _Float16 v16h;
typedef __attribute__((ext_vector_type(8)))  _Float16 v8h;
typedef __attribute__((ext_vector_type(8)))  float    v8f;

#define F_IN  256
#define F_OUT 64
#define WT_STRIDE 264   // halves per row: 528B, 16B aligned, bank-stagger
#define ALPHA 0.2f

// ---------------------------------------------------------------------------
// Kernel 0: zero-init output accumulator and rowsum (must run every call).
// ---------------------------------------------------------------------------
__global__ void gat_init_kernel(float* __restrict__ out, float* __restrict__ rowsum, int N) {
    int t = blockIdx.x * blockDim.x + threadIdx.x;
    if (t < N * F_OUT) out[t] = 0.0f;
    if (t < N)         rowsum[t] = 0.0f;
}

// ---------------------------------------------------------------------------
// Kernel 1: Wh = x @ W (f16 WMMA, f32 accum) + fused s_src/s_dst epilogue.
// Block = 256 threads = 8 waves; each wave computes a 16x64 tile of Wh.
// ---------------------------------------------------------------------------
__device__ __forceinline__ v16h load_b_frag(const _Float16* bp) {
    v8h lo = *(const v8h*)(bp);
    v8h hi = *(const v8h*)(bp + 8);
    return __builtin_shufflevector(lo, hi, 0,1,2,3,4,5,6,7,8,9,10,11,12,13,14,15);
}

__global__ void __launch_bounds__(256)
gat_gemm_kernel(const float* __restrict__ x, const float* __restrict__ W,
                const float* __restrict__ a,
                float* __restrict__ Wh, float* __restrict__ s_src,
                float* __restrict__ s_dst, int N) {
    __shared__ _Float16 Wt[F_OUT * WT_STRIDE];   // W transposed: Wt[n][k], 33.8 KB

    const int tid = threadIdx.x;

    // Stage W^T into LDS as f16 (once per block). Reads coalesced.
    for (int idx = tid; idx < F_IN * F_OUT; idx += 256) {
        int k = idx >> 6;        // row of W
        int n = idx & 63;        // col of W
        Wt[n * WT_STRIDE + k] = (_Float16)W[idx];
    }
    __syncthreads();

    const int wave = tid >> 5;
    const int lane = tid & 31;
    const int hh   = lane >> 4;   // half-wave: 0 or 1
    const int lr   = lane & 15;

    const int row_base = blockIdx.x * 128 + wave * 16;
    int mA = row_base + lr;                 // A-matrix row owned by this lane
    if (mA > N - 1) mA = N - 1;             // clamp: WMMA needs EXEC all-ones
    const float* xrow = x + (size_t)mA * F_IN;

    // Base LDS pointer for this lane's B-fragment column slice.
    const _Float16* bbase = &Wt[lr * WT_STRIDE + 16 * hh];

    v8f acc0 = {}, acc1 = {}, acc2 = {}, acc3 = {};

    #pragma unroll
    for (int kb = 0; kb < 8; ++kb) {
        const int k0 = kb * 32;
        // ---- A fragment: contiguous k order, same as the B fragments.
        // half 0 lanes hold K = {k0..k0+15}
        // half 1 lanes hold K = {k0+16..k0+31}
        const float4* p0 = (const float4*)(xrow + k0 + 16 * hh);
        const float4* p1 = (const float4*)(xrow + k0 + 16 * hh + 8);
        float4 f0 = p0[0], f1 = p0[1];
        float4 f2 = p1[0], f3 = p1[1];
        __builtin_prefetch(xrow + k0 + 128, 0, 3);  // global_prefetch_b8, 4 k-steps ahead
        v16h A;
        A[0]=(_Float16)f0.x; A[1]=(_Float16)f0.y; A[2]=(_Float16)f0.z; A[3]=(_Float16)f0.w;
        A[4]=(_Float16)f1.x; A[5]=(_Float16)f1.y; A[6]=(_Float16)f1.z; A[7]=(_Float16)f1.w;
        A[8]=(_Float16)f2.x; A[9]=(_Float16)f2.y; A[10]=(_Float16)f2.z; A[11]=(_Float16)f2.w;
        A[12]=(_Float16)f3.x; A[13]=(_Float16)f3.y; A[14]=(_Float16)f3.z; A[15]=(_Float16)f3.w;

        // ---- B fragments: lane (half hh, lr) holds column 16j+lr,
        // K = k0 + 16*hh .. +15 (16 contiguous halves in Wt).
        // Load all four BEFORE the WMMA quartet so the ds_loads overlap WMMA
        // issue (staggered s_wait_dscnt instead of load->wait->wmma chains).
        const _Float16* bk = bbase + k0;
        v16h B0 = load_b_frag(bk);
        v16h B1 = load_b_frag(bk + 16 * WT_STRIDE);
        v16h B2 = load_b_frag(bk + 32 * WT_STRIDE);
        v16h B3 = load_b_frag(bk + 48 * WT_STRIDE);

        acc0 = __builtin_amdgcn_wmma_f32_16x16x32_f16(false, A, false, B0, (short)0, acc0, false, false);
        acc1 = __builtin_amdgcn_wmma_f32_16x16x32_f16(false, A, false, B1, (short)0, acc1, false, false);
        acc2 = __builtin_amdgcn_wmma_f32_16x16x32_f16(false, A, false, B2, (short)0, acc2, false, false);
        acc3 = __builtin_amdgcn_wmma_f32_16x16x32_f16(false, A, false, B3, (short)0, acc3, false, false);
    }

    // Attention vector slices for this lane's columns (n = 16j + lr).
    float asv[4], adv[4];
    #pragma unroll
    for (int j = 0; j < 4; ++j) {
        asv[j] = a[16 * j + lr];
        adv[j] = a[F_OUT + 16 * j + lr];
    }

    // C/D layout: VGPR r, lane l -> M = r + 8*(l/16), N = (l%16) + 16j.
    #pragma unroll
    for (int r = 0; r < 8; ++r) {
        const int row = row_base + r + 8 * hh;
        const float c0 = acc0[r], c1 = acc1[r], c2 = acc2[r], c3 = acc3[r];
        if (row < N) {
            float* wr = Wh + (size_t)row * F_OUT;
            wr[lr]      = c0;
            wr[lr + 16] = c1;
            wr[lr + 32] = c2;
            wr[lr + 48] = c3;
        }
        // Fused dot with attention vectors; reduce across the 16 lanes of the half.
        float ps = c0 * asv[0] + c1 * asv[1] + c2 * asv[2] + c3 * asv[3];
        float pd = c0 * adv[0] + c1 * adv[1] + c2 * adv[2] + c3 * adv[3];
        #pragma unroll
        for (int m = 8; m >= 1; m >>= 1) {
            ps += __shfl_xor(ps, m, 32);
            pd += __shfl_xor(pd, m, 32);
        }
        if (lr == 0 && row < N) {
            s_src[row] = ps;
            s_dst[row] = pd;
        }
    }
}

// ---------------------------------------------------------------------------
// Kernel 2: per-edge score -> exp(-leaky_relu) + rowsum atomic accumulation.
// ---------------------------------------------------------------------------
__global__ void gat_edge_kernel(const int* __restrict__ edge,
                                const float* __restrict__ s_src,
                                const float* __restrict__ s_dst,
                                float* __restrict__ edge_e,
                                float* __restrict__ rowsum, int E) {
    int e = blockIdx.x * blockDim.x + threadIdx.x;
    if (e >= E) return;
    int s = edge[e];
    int d = edge[E + e];
    float sc = s_src[s] + s_dst[d];
    float lrl = sc > 0.0f ? sc : ALPHA * sc;
    float ee = __expf(-lrl);
    edge_e[e] = ee;
    unsafeAtomicAdd(rowsum + s, ee);
}

// ---------------------------------------------------------------------------
// Kernel 3: scatter aggregation. One wave per edge; lane handles 2 features.
// Wh and the accumulator are L2-resident (12.8 MB each vs 192 MB L2).
// ---------------------------------------------------------------------------
__global__ void __launch_bounds__(256)
gat_agg_kernel(const int* __restrict__ edge, const float* __restrict__ edge_e,
               const float* __restrict__ Wh, float* __restrict__ out, int E) {
    int gwave = (blockIdx.x * 256 + threadIdx.x) >> 5;
    int lane  = threadIdx.x & 31;
    if (gwave >= E) return;
    int s = edge[gwave];
    int d = edge[E + gwave];
    float ee = edge_e[gwave];
    const float* wr = Wh + (size_t)d * F_OUT;
    float w0 = wr[lane];
    float w1 = wr[lane + 32];
    float* orow = out + (size_t)s * F_OUT;
    unsafeAtomicAdd(orow + lane,      ee * w0);
    unsafeAtomicAdd(orow + lane + 32, ee * w1);
}

// ---------------------------------------------------------------------------
// Kernel 4: h' / rowsum, then ELU, in place on d_out.
// ---------------------------------------------------------------------------
__global__ void gat_final_kernel(float* __restrict__ out,
                                 const float* __restrict__ rowsum, int N) {
    int t = blockIdx.x * blockDim.x + threadIdx.x;
    if (t >= N * F_OUT) return;
    float v = out[t] / rowsum[t >> 6];
    out[t] = v > 0.0f ? v : (__expf(v) - 1.0f);
}

// ---------------------------------------------------------------------------
extern "C" void kernel_launch(void* const* d_in, const int* in_sizes, int n_in,
                              void* d_out, int out_size, void* d_ws, size_t ws_size,
                              hipStream_t stream) {
    const float* x    = (const float*)d_in[0];
    const int*   edge = (const int*)  d_in[1];
    const float* W    = (const float*)d_in[2];
    const float* a    = (const float*)d_in[3];
    float*       out  = (float*)d_out;

    const int N = in_sizes[0] / F_IN;
    const int E = in_sizes[1] / 2;

    // Workspace carve-out.
    float* Wh     = (float*)d_ws;          // N * 64
    float* s_src  = Wh + (size_t)N * F_OUT;
    float* s_dst  = s_src + N;
    float* rowsum = s_dst + N;
    float* edge_e = rowsum + N;            // E

    // 0) zero accumulators (graph replays reuse buffers).
    {
        int total = N * F_OUT;
        gat_init_kernel<<<(total + 255) / 256, 256, 0, stream>>>(out, rowsum, N);
    }
    // 1) WMMA GEMM + fused score epilogue.
    {
        int grid = (N + 127) / 128;
        gat_gemm_kernel<<<grid, 256, 0, stream>>>(x, W, a, Wh, s_src, s_dst, N);
    }
    // 2) edge scores.
    gat_edge_kernel<<<(E + 255) / 256, 256, 0, stream>>>(edge, s_src, s_dst, edge_e, rowsum, E);
    // 3) scatter aggregation (1 wave per edge).
    {
        long long threads = (long long)E * 32;
        int grid = (int)((threads + 255) / 256);
        gat_agg_kernel<<<grid, 256, 0, stream>>>(edge, edge_e, Wh, out, E);
    }
    // 4) normalize + ELU.
    gat_final_kernel<<<(N * F_OUT + 255) / 256, 256, 0, stream>>>(out, rowsum, N);
}
